// Model_EdgeCycle_old_50869592655569
// MI455X (gfx1250) — matrix operand order repaired and checked by reading, not compiled, not run on hardware
//
#include <hip/hip_runtime.h>
#include <stdint.h>

// ---------------------------------------------------------------------------
// Static problem sizes (from the reference)
// ---------------------------------------------------------------------------
#define TPB 256
static constexpr long kN    = 50000;
static constexpr long kE    = 100000;
static constexpr long kC5   = 10000;
static constexpr long kC6   = 10000;
static constexpr long kB    = 500;
static constexpr long kR5   = 5 * kC5;     // 50000
static constexpr long kR6   = 6 * kC6;     // 60000
static constexpr long kRC   = kR5 + kR6;   // 110000
static constexpr long kTwoE = 2 * kE;      // 200000
static constexpr long kBPad = 512;         // B padded to multiple of 16

typedef __attribute__((ext_vector_type(16))) __bf16 v16bf;
typedef __attribute__((ext_vector_type(8)))  float  v8f;

__device__ __forceinline__ uint16_t f2bf(float f) {
  uint32_t u = __float_as_uint(f);
  u += 0x7FFFu + ((u >> 16) & 1u);   // round-to-nearest-even
  return (uint16_t)(u >> 16);
}

static inline int nblk(long n) { return (int)((n + TPB - 1) / TPB); }

// ---------------------------------------------------------------------------
// Elementwise / index / gather / scatter kernels
// ---------------------------------------------------------------------------
__global__ void k_zero_f32(float* p, long n) {
  long i = blockIdx.x * (long)blockDim.x + threadIdx.x;
  if (i < n) p[i] = 0.0f;
}
__global__ void k_zero_u16(uint16_t* p, long n) {
  long i = blockIdx.x * (long)blockDim.x + threadIdx.x;
  if (i < n) p[i] = 0;
}
__global__ void k_copy_i32(int* dst, const int* src, long n) {
  long i = blockIdx.x * (long)blockDim.x + threadIdx.x;
  if (i < n) dst[i] = src[i];
}
__global__ void k_iota_div(int* p, long n, int div, int off) {
  long i = blockIdx.x * (long)blockDim.x + threadIdx.x;
  if (i < n) p[i] = (int)(i / div) + off;
}
__global__ void k_build_ea(const int* ei, int* ea, long E) {
  long i = blockIdx.x * (long)blockDim.x + threadIdx.x;
  if (i < E) { ea[2 * i] = ei[i]; ea[2 * i + 1] = ei[E + i]; }
}

// dst[dIds[r]*dstStride + colOff + c] += src[sIds[r]*C + c]   (ids nullable->identity)
__global__ void k_scat(float* dst, long dstStride, long colOff,
                       const float* __restrict__ src,
                       const int* dIds, const int* sIds, long R, int C) {
  long idx = blockIdx.x * (long)blockDim.x + threadIdx.x;
  long tot = R * (long)C;
  if (idx >= tot) return;
  long r = idx / C; int c = (int)(idx - r * C);
  long dr = dIds ? (long)dIds[r] : r;
  long sr = sIds ? (long)sIds[r] : r;
  atomicAdd(&dst[dr * dstStride + colOff + c], src[sr * (long)C + c]);
}

__global__ void k_gath_f32(float* dst, long dstStride, long colOff,
                           const float* __restrict__ src, const int* ids, long R, int C) {
  long idx = blockIdx.x * (long)blockDim.x + threadIdx.x;
  long tot = R * (long)C;
  if (idx >= tot) return;
  long r = idx / C; int c = (int)(idx - r * C);
  long sr = ids ? (long)ids[r] : r;
  dst[r * dstStride + colOff + c] = src[sr * (long)C + c];
}

__global__ void k_gath_bf16(uint16_t* dst, long dstStride, long colOff,
                            const float* __restrict__ src, const int* ids, long R, int C) {
  long idx = blockIdx.x * (long)blockDim.x + threadIdx.x;
  long tot = R * (long)C;
  if (idx >= tot) return;
  long r = idx / C; int c = (int)(idx - r * C);
  long sr = ids ? (long)ids[r] : r;
  dst[r * dstStride + colOff + c] = f2bf(src[sr * (long)C + c]);
}

__global__ void k_bias_relu_bf16(uint16_t* dst, long dstStride, long colOff,
                                 const float* __restrict__ x, const float* bias, long M, int C) {
  long idx = blockIdx.x * (long)blockDim.x + threadIdx.x;
  long tot = M * (long)C;
  if (idx >= tot) return;
  long r = idx / C; int c = (int)(idx - r * C);
  float v = x[r * (long)C + c] + bias[c];
  dst[r * dstStride + colOff + c] = f2bf(v > 0.0f ? v : 0.0f);
}
__global__ void k_bias_relu_f32(float* dst, const float* __restrict__ x, const float* bias,
                                long M, int C) {
  long idx = blockIdx.x * (long)blockDim.x + threadIdx.x;
  long tot = M * (long)C;
  if (idx >= tot) return;
  long r = idx / C; int c = (int)(idx - r * C);
  float v = x[r * (long)C + c] + bias[c];
  dst[idx] = v > 0.0f ? v : 0.0f;
}

// BatchNorm column statistics (one block per column, LDS tree reduce)
__global__ void k_colstats(const float* __restrict__ x, long M, int C, float* mean, float* rstd) {
  int c = blockIdx.x;
  float s = 0.0f, ss = 0.0f;
  for (long r = threadIdx.x; r < M; r += blockDim.x) {
    float v = x[r * (long)C + c];
    s += v; ss += v * v;
  }
  __shared__ float sh1[TPB], sh2[TPB];
  sh1[threadIdx.x] = s; sh2[threadIdx.x] = ss;
  __syncthreads();
  for (int st = TPB / 2; st > 0; st >>= 1) {
    if ((int)threadIdx.x < st) {
      sh1[threadIdx.x] += sh1[threadIdx.x + st];
      sh2[threadIdx.x] += sh2[threadIdx.x + st];
    }
    __syncthreads();
  }
  if (threadIdx.x == 0) {
    float m   = sh1[0] / (float)M;
    float var = sh2[0] / (float)M - m * m;
    mean[c] = m;
    rstd[c] = rsqrtf(fmaxf(var, 0.0f) + 1e-5f);
  }
}

__global__ void k_bn_relu_bf16(uint16_t* dst, long dstStride, long colOff,
                               const float* __restrict__ x, const float* mean, const float* rstd,
                               const float* g, const float* b, long M, int C) {
  long idx = blockIdx.x * (long)blockDim.x + threadIdx.x;
  long tot = M * (long)C;
  if (idx >= tot) return;
  long r = idx / C; int c = (int)(idx - r * C);
  float v = g[c] * (x[r * (long)C + c] - mean[c]) * rstd[c] + b[c];
  dst[r * dstStride + colOff + c] = f2bf(v > 0.0f ? v : 0.0f);
}
__global__ void k_bn_relu_f32(float* dst, const float* __restrict__ x,
                              const float* mean, const float* rstd,
                              const float* g, const float* b, long M, int C) {
  long idx = blockIdx.x * (long)blockDim.x + threadIdx.x;
  long tot = M * (long)C;
  if (idx >= tot) return;
  long r = idx / C; int c = (int)(idx - r * C);
  float v = g[c] * (x[r * (long)C + c] - mean[c]) * rstd[c] + b[c];
  dst[idx] = v > 0.0f ? v : 0.0f;
}

// ---------------------------------------------------------------------------
// WMMA bf16 GEMM:  C[M,Nc] = A[M,K](bf16) @ W[K,Nc]
// Weights pre-packed per 16x16x32 fragment so each lane's B fragment is
// 16 contiguous bf16 values (two b128 loads).
// Packed layout flat index: (((nt*Kt + t)*2 + h)*16 + n)*16 + e
//   where k = t*32 + h*16 + e, col = nt*16 + n
// ---------------------------------------------------------------------------
__global__ void k_packw(const float* __restrict__ W, uint16_t* __restrict__ Wp, int K, int Nc) {
  long idx = blockIdx.x * (long)blockDim.x + threadIdx.x;
  long tot = (long)K * Nc;
  if (idx >= tot) return;
  int kt = K >> 5;
  int e  = (int)(idx & 15);
  int n  = (int)((idx >> 4) & 15);
  int h  = (int)((idx >> 8) & 1);
  long rest = idx >> 9;
  int t  = (int)(rest % kt);
  int nt = (int)(rest / kt);
  int k   = t * 32 + h * 16 + e;
  int col = nt * 16 + n;
  Wp[idx] = f2bf(W[(long)k * Nc + col]);
}

// Each block: 8 waves x (one 16-row M tile each) x 2 N tiles (dual accumulators).
// The two N-tile weight slabs (contiguous in the packed layout) are staged into
// LDS once per block with GLOBAL_LOAD_ASYNC_TO_LDS_B128 (ASYNCcnt), then each
// k-step does: 2x b128 A loads (global) + 4x b128 B loads (LDS) + 2x WMMA.
#define MAXKT 22   // K <= 704
__global__ __launch_bounds__(256)
void k_gemm_bf16(const uint16_t* __restrict__ A, const uint16_t* __restrict__ Wp,
                 float* __restrict__ C, int Mtiles, int K, int Nc) {
  __shared__ __align__(16) uint16_t sw[2 * MAXKT * 512];   // 45 KB: two N-tile slabs

  int tid  = threadIdx.x;
  int lane = tid & 31;
  int wave = tid >> 5;
  int kt   = K >> 5;
  int ntile0 = blockIdx.y * 2;

  // ---- cooperative async copy: global packed weights -> LDS ----
  {
    const uint16_t* slab = Wp + ((long)ntile0 * kt) * 512;     // 2*kt*512 bf16, contiguous
    int chunks = 2 * kt * 64;                                  // 16-byte chunks
    __attribute__((address_space(3))) uint16_t* lb =
        (__attribute__((address_space(3))) uint16_t*)sw;
    for (int i = tid; i < chunks; i += 256) {                  // exact multiple of 256: no divergence
      uint32_t laddr = (uint32_t)(uintptr_t)(lb + (long)i * 8);
      uint64_t gaddr = (uint64_t)(uintptr_t)(slab + (long)i * 8);
      asm volatile("global_load_async_to_lds_b128 %0, %1, off"
                   :: "v"(laddr), "v"(gaddr)
                   : "memory");
    }
    asm volatile("s_wait_asynccnt 0x0" ::: "memory");
  }
  __syncthreads();   // all threads reach the barrier before any tail-wave exit

  int mtile = blockIdx.x * 8 + wave;
  if (mtile >= Mtiles) return;          // wave-uniform: EXEC stays all-ones for WMMA

  int n = lane & 15;                     // B column within tile / A row within tile
  int h = lane >> 4;                     // lane-half selects K sub-range

  long row = (long)mtile * 16 + n;       // A-matrix: lanes 0-15 and 16-31 both hold M=0..15
  const uint16_t* arow = A + row * (long)K + h * 16;
  const uint16_t* bl0  = sw + lane * 16;               // slab 0
  const uint16_t* bl1  = sw + (long)kt * 512 + lane * 16;  // slab 1

  v8f acc0 = {0.f, 0.f, 0.f, 0.f, 0.f, 0.f, 0.f, 0.f};
  v8f acc1 = {0.f, 0.f, 0.f, 0.f, 0.f, 0.f, 0.f, 0.f};
  for (int t = 0; t < kt; ++t) {
    union { uint4 q[2]; v16bf v; } a, b0, b1;
    // A fragment: K = t*32 + h*16 + {0..15}
    a.q[0] = *(const uint4*)(arow + (long)t * 32);
    a.q[1] = *(const uint4*)(arow + (long)t * 32 + 8);
    // B fragments from LDS (ds_load_b128)
    b0.q[0] = *(const uint4*)(bl0 + (long)t * 512);
    b0.q[1] = *(const uint4*)(bl0 + (long)t * 512 + 8);
    b1.q[0] = *(const uint4*)(bl1 + (long)t * 512);
    b1.q[1] = *(const uint4*)(bl1 + (long)t * 512 + 8);
    if (t + 1 < kt) __builtin_prefetch(arow + (long)(t + 1) * 32, 0, 3);
    acc0 = __builtin_amdgcn_wmma_f32_16x16x32_bf16(false, a.v, false, b0.v,
                                                   (short)0, acc0, false, false);
    acc1 = __builtin_amdgcn_wmma_f32_16x16x32_bf16(false, a.v, false, b1.v,
                                                   (short)0, acc1, false, false);
  }
  // C layout: VGPR r -> row = mtile*16 + h*8 + r, col = ntile*16 + n
  float* crow = C + ((long)mtile * 16 + h * 8) * (long)Nc + (long)ntile0 * 16 + n;
#pragma unroll
  for (int r = 0; r < 8; ++r) {
    crow[(long)r * Nc]      = acc0[r];
    crow[(long)r * Nc + 16] = acc1[r];
  }
}

// ---------------------------------------------------------------------------
// Final pooling / head kernels
// ---------------------------------------------------------------------------
__global__ void k_count_batch(const int* batch, float* cnt, long n) {
  long i = blockIdx.x * (long)blockDim.x + threadIdx.x;
  if (i < n) atomicAdd(&cnt[batch[i]], 1.0f);
}
__global__ void k_pool_div_bf16(const float* __restrict__ g, const float* cnt,
                                uint16_t* X, long Bn, int C) {
  long idx = blockIdx.x * (long)blockDim.x + threadIdx.x;
  long tot = Bn * (long)C;
  if (idx >= tot) return;
  long r = idx / C; int c = (int)(idx - r * C);
  float d = fmaxf(cnt[r], 1.0f);
  X[r * (long)C + c] = f2bf(g[r * (long)C + c] / d);
}
__global__ void k_outhead(const float* __restrict__ h, const float* __restrict__ w,
                          const float* b, float* out, long Bn, int C) {
  long i = blockIdx.x * (long)blockDim.x + threadIdx.x;
  if (i >= Bn) return;
  float s = b[0];
  for (int c = 0; c < C; ++c) s += h[i * (long)C + c] * w[c];
  out[i] = s;
}

// ---------------------------------------------------------------------------
// Host-side helpers
// ---------------------------------------------------------------------------
static void run_gemm(hipStream_t s, const uint16_t* A, long M, int K, int Nc,
                     const float* W, uint16_t* Wp, float* C) {
  long tot = (long)K * Nc;
  k_packw<<<nblk(tot), TPB, 0, s>>>(W, Wp, K, Nc);
  int Mt = (int)(M / 16);
  dim3 grid((Mt + 7) / 8, Nc / 32);    // 2 N-tiles per block
  k_gemm_bf16<<<grid, dim3(256), 0, s>>>(A, Wp, C, Mt, K, Nc);
}

struct G11 {
  hipStream_t s;
  float *tmpL, *tmpB, *tsum;  // tmpL/tmpB: [kN, C]; tsum: [n_t, C]
  int C;
};

// Scatter phase of gather11: local = seg(rows,atoms,N); tsum = seg(rows,tid,n_t);
// bc[atoms[r]] += tsum[tid[r]]
static void g11_scatter(G11& g, const float* rows, const int* sIds,
                        const int* atoms, const int* tid, long R, long n_t, int C) {
  g.C = C;
  long nc = kN * (long)C;
  k_zero_f32<<<nblk(nc), TPB, 0, g.s>>>(g.tmpL, nc);
  k_zero_f32<<<nblk(nc), TPB, 0, g.s>>>(g.tmpB, nc);
  long tc = n_t * (long)C;
  k_zero_f32<<<nblk(tc), TPB, 0, g.s>>>(g.tsum, tc);
  long rc = R * (long)C;
  k_scat<<<nblk(rc), TPB, 0, g.s>>>(g.tmpL, C, 0, rows, atoms, sIds, R, C);
  k_scat<<<nblk(rc), TPB, 0, g.s>>>(g.tsum, C, 0, rows, tid,   sIds, R, C);
  k_scat<<<nblk(rc), TPB, 0, g.s>>>(g.tmpB, C, 0, g.tsum, atoms, tid, R, C);
}
static void g11_gather_f32(G11& g, float* dst, long stride, long off, const int* tgt, long Rt) {
  long rc = Rt * (long)g.C;
  k_gath_f32<<<nblk(rc), TPB, 0, g.s>>>(dst, stride, off,        g.tmpL, tgt, Rt, g.C);
  k_gath_f32<<<nblk(rc), TPB, 0, g.s>>>(dst, stride, off + g.C,  g.tmpB, tgt, Rt, g.C);
}
static void g11_gather_bf16(G11& g, uint16_t* dst, long stride, long off, const int* tgt, long Rt) {
  long rc = Rt * (long)g.C;
  k_gath_bf16<<<nblk(rc), TPB, 0, g.s>>>(dst, stride, off,       g.tmpL, tgt, Rt, g.C);
  k_gath_bf16<<<nblk(rc), TPB, 0, g.s>>>(dst, stride, off + g.C, g.tmpB, tgt, Rt, g.C);
}

// mlp2: Linear(K->128,no bias) -> BN -> ReLU -> Linear(128->64,no bias) -> BN -> ReLU
static void run_mlp2(hipStream_t s, const uint16_t* X, long M, int K,
                     const float* w1, const float* g1, const float* b1,
                     const float* w2, const float* g2, const float* b2,
                     uint16_t* Xi, float* G1, float* G2, uint16_t* Wp, float* stat,
                     uint16_t* dstB, long dstStride, long dstOff, float* dstF) {
  run_gemm(s, X, M, K, 128, w1, Wp, G1);
  k_colstats<<<128, TPB, 0, s>>>(G1, M, 128, stat, stat + 256);
  k_bn_relu_bf16<<<nblk(M * 128), TPB, 0, s>>>(Xi, 128, 0, G1, stat, stat + 256, g1, b1, M, 128);
  run_gemm(s, Xi, M, 128, 64, w2, Wp, G2);
  k_colstats<<<64, TPB, 0, s>>>(G2, M, 64, stat, stat + 256);
  if (dstB)
    k_bn_relu_bf16<<<nblk(M * 64), TPB, 0, s>>>(dstB, dstStride, dstOff, G2, stat, stat + 256, g2, b2, M, 64);
  else
    k_bn_relu_f32<<<nblk(M * 64), TPB, 0, s>>>(dstF, G2, stat, stat + 256, g2, b2, M, 64);
}

// ---------------------------------------------------------------------------
// kernel_launch
//
// Input order assumption: top-level setup_inputs() dict order
//   [0]x [1]edge_attr [2]edge_index [3]c5_atoms [4]c6_atoms [5]batch,
// followed by the 'params' pytree leaves in JAX sorted-key order:
//   cyc_b1,cyc_b2,cyc_w1,cyc_w2, edge_b1,edge_b2,edge_emb,edge_w1,edge_w2,
//   fin_b1,fin_b2,fin_bb1,fin_bb2,fin_g1,fin_g2,fin_w1,fin_w2,
//   layers[0..1]{ecC_b1,ecC_b2,ecC_g1,ecC_g2,ecC_w1,ecC_w2,
//                ecE_b1,ecE_b2,ecE_g1,ecE_g2,ecE_w1,ecE_w2,
//                ee_b1,ee_b2,ee_g1,ee_g2,ee_w1,ee_w2,mix_b,mix_g,mix_w},
//   node_emb, out_b, out_w
// ---------------------------------------------------------------------------
extern "C" void kernel_launch(void* const* d_in, const int* in_sizes, int n_in,
                              void* d_out, int out_size, void* d_ws, size_t ws_size,
                              hipStream_t stream) {
  (void)in_sizes; (void)n_in; (void)out_size; (void)ws_size;
  hipStream_t s = stream;

  const int* x          = (const int*)d_in[0];
  const int* edge_attr  = (const int*)d_in[1];
  const int* edge_index = (const int*)d_in[2];
  const int* c5a        = (const int*)d_in[3];
  const int* c6a        = (const int*)d_in[4];
  const int* batch      = (const int*)d_in[5];

  const float* cyc_b1  = (const float*)d_in[6];
  const float* cyc_b2  = (const float*)d_in[7];
  const float* cyc_w1  = (const float*)d_in[8];
  const float* cyc_w2  = (const float*)d_in[9];
  const float* edge_b1 = (const float*)d_in[10];
  const float* edge_b2 = (const float*)d_in[11];
  const float* edge_emb= (const float*)d_in[12];
  const float* edge_w1 = (const float*)d_in[13];
  const float* edge_w2 = (const float*)d_in[14];
  // d_in[15]=fin_b1, d_in[16]=fin_b2 : biases before BN cancel out -> unused
  const float* fin_bb1 = (const float*)d_in[17];
  const float* fin_bb2 = (const float*)d_in[18];
  const float* fin_g1  = (const float*)d_in[19];
  const float* fin_g2  = (const float*)d_in[20];
  const float* fin_w1  = (const float*)d_in[21];
  const float* fin_w2  = (const float*)d_in[22];
  const float* LP[2][21];
  for (int l = 0; l < 2; ++l)
    for (int j = 0; j < 21; ++j) LP[l][j] = (const float*)d_in[23 + l * 21 + j];
  const float* node_emb = (const float*)d_in[65];
  const float* out_b    = (const float*)d_in[66];
  const float* out_w    = (const float*)d_in[67];

  // ---- workspace bump allocator (deterministic) ----
  char* base = (char*)d_ws;
  size_t off = 0;
  auto alloc = [&](size_t bytes) -> void* {
    void* p = base + off;
    off = (off + bytes + 255) & ~(size_t)255;
    return p;
  };
  int* ea     = (int*)alloc(kTwoE * 4);
  int* etid   = (int*)alloc(kTwoE * 4);
  int* c5t    = (int*)alloc(kR5 * 4);
  int* c6t    = (int*)alloc(kR6 * 4);
  int* cyca   = (int*)alloc(kRC * 4);
  int* cyct   = (int*)alloc(kRC * 4);
  float* nf     = (float*)alloc(kN * 64 * 4);
  float* ef0    = (float*)alloc(kE * 64 * 4);
  float* er     = (float*)alloc(kTwoE * 64 * 4);
  float* cyc    = (float*)alloc(kRC * 64 * 4);
  float* tmpL   = (float*)alloc(kN * 320 * 4);
  float* tmpB   = (float*)alloc(kN * 320 * 4);
  float* tsum   = (float*)alloc(kE * 128 * 4);
  float* ee1    = (float*)alloc(kTwoE * 128 * 4);
  float* ecT    = (float*)alloc(kRC * 128 * 4);
  float* G1     = (float*)alloc(kTwoE * 128 * 4);
  float* G2     = (float*)alloc(kTwoE * 64 * 4);
  float* cycnew = (float*)alloc(kRC * 320 * 4);
  uint16_t* Xa  = (uint16_t*)alloc(kTwoE * 320 * 2);  // X_ee, then X_ecC
  uint16_t* Xb  = (uint16_t*)alloc(kTwoE * 704 * 2);  // X0/Xcyc, then X_ecE
  uint16_t* Xc  = (uint16_t*)alloc(kTwoE * 128 * 2);  // Xmix, fin stage-2 input
  uint16_t* Xi  = (uint16_t*)alloc(kTwoE * 128 * 2);  // mlp2 intermediates, fin stage-1 input
  uint16_t* Wp  = (uint16_t*)alloc(704 * 128 * 2);
  float* stat   = (float*)alloc(1024 * 4);            // mean @0, rstd @256
  float* cnt    = (float*)alloc(512 * 4);
  float* gp     = (float*)alloc(512 * 128 * 4);

  G11 g11{ s, tmpL, tmpB, tsum, 0 };

  // ---- index arrays ----
  k_build_ea<<<nblk(kE), TPB, 0, s>>>(edge_index, ea, kE);
  k_iota_div<<<nblk(kTwoE), TPB, 0, s>>>(etid, kTwoE, 2, 0);
  k_iota_div<<<nblk(kR5), TPB, 0, s>>>(c5t, kR5, 5, 0);
  k_iota_div<<<nblk(kR6), TPB, 0, s>>>(c6t, kR6, 6, 0);
  k_copy_i32<<<nblk(kR5), TPB, 0, s>>>(cyca, c5a, kR5);
  k_copy_i32<<<nblk(kR6), TPB, 0, s>>>(cyca + kR5, c6a, kR6);
  k_iota_div<<<nblk(kR5), TPB, 0, s>>>(cyct, kR5, 5, 0);
  k_iota_div<<<nblk(kR6), TPB, 0, s>>>(cyct + kR5, kR6, 6, (int)kC5);

  // ---- embeddings ----
  k_gath_f32<<<nblk(kN * 64), TPB, 0, s>>>(nf, 64, 0, node_emb, x, kN, 64);
  k_gath_f32<<<nblk(kE * 64), TPB, 0, s>>>(ef0, 64, 0, edge_emb, edge_attr, kE, 64);

  // ---- initial edge representation ----
  k_zero_f32<<<nblk(kN * 64), TPB, 0, s>>>(tmpL, kN * 64);
  k_scat<<<nblk(kTwoE * 64), TPB, 0, s>>>(tmpL, 64, 0, ef0, ea, etid, kTwoE, 64); // seg(ef0[etid], ea, N)
  k_gath_bf16<<<nblk(kTwoE * 64), TPB, 0, s>>>(Xb, 128, 0,  nf,   ea, kTwoE, 64);
  k_gath_bf16<<<nblk(kTwoE * 64), TPB, 0, s>>>(Xb, 128, 64, tmpL, ea, kTwoE, 64);
  run_gemm(s, Xb, kTwoE, 128, 128, edge_w1, Wp, G1);
  k_bias_relu_bf16<<<nblk(kTwoE * 128), TPB, 0, s>>>(Xi, 128, 0, G1, edge_b1, kTwoE, 128);
  run_gemm(s, Xi, kTwoE, 128, 64, edge_w2, Wp, G2);
  k_bias_relu_f32<<<nblk(kTwoE * 64), TPB, 0, s>>>(er, G2, edge_b2, kTwoE, 64);

  // ---- initial cycle representation ----
  g11_scatter(g11, nf, c5a, c5a, c5t, kR5, kC5, 64);
  g11_gather_bf16(g11, Xb, 128, 0, c5a, kR5);
  g11_scatter(g11, nf, c6a, c6a, c6t, kR6, kC6, 64);
  g11_gather_bf16(g11, Xb + (size_t)kR5 * 128, 128, 0, c6a, kR6);
  run_gemm(s, Xb, kRC, 128, 128, cyc_w1, Wp, G1);
  k_bias_relu_bf16<<<nblk(kRC * 128), TPB, 0, s>>>(Xi, 128, 0, G1, cyc_b1, kRC, 128);
  run_gemm(s, Xi, kRC, 128, 64, cyc_w2, Wp, G2);
  k_bias_relu_f32<<<nblk(kRC * 64), TPB, 0, s>>>(cyc, G2, cyc_b2, kRC, 64);

  // ---- layers ----
  for (int l = 0; l < 2; ++l) {
    const float* ecC_b1 = LP[l][0];  const float* ecC_b2 = LP[l][1];
    const float* ecC_g1 = LP[l][2];  const float* ecC_g2 = LP[l][3];
    const float* ecC_w1 = LP[l][4];  const float* ecC_w2 = LP[l][5];
    const float* ecE_b1 = LP[l][6];  const float* ecE_b2 = LP[l][7];
    const float* ecE_g1 = LP[l][8];  const float* ecE_g2 = LP[l][9];
    const float* ecE_w1 = LP[l][10]; const float* ecE_w2 = LP[l][11];
    const float* ee_b1  = LP[l][12]; const float* ee_b2  = LP[l][13];
    const float* ee_g1  = LP[l][14]; const float* ee_g2  = LP[l][15];
    const float* ee_w1  = LP[l][16]; const float* ee_w2  = LP[l][17];
    const float* mix_b  = LP[l][18]; const float* mix_g  = LP[l][19];
    const float* mix_w  = LP[l][20];

    // Shared first-stage gather11 of er over (ea, etid): reused by ee, ec5, ec6
    g11_scatter(g11, er, nullptr, ea, etid, kTwoE, kE, 64);
    g11_gather_f32(g11, ee1, 128, 0, ea, kTwoE);                       // ee stage 1 [2E,128]
    g11_gather_f32(g11, ecT, 128, 0, c5a, kR5);                        // ec5 [5C5,128]
    g11_gather_f32(g11, ecT + (size_t)kR5 * 128, 128, 0, c6a, kR6);    // ec6 [6C6,128]

    // Edge_edge: second gather11 -> X_ee = [er | gather11(ee1)] in bf16 [2E,320]
    k_gath_bf16<<<nblk(kTwoE * 64), TPB, 0, s>>>(Xa, 320, 0, er, nullptr, kTwoE, 64);
    g11_scatter(g11, ee1, nullptr, ea, etid, kTwoE, kE, 128);
    g11_gather_bf16(g11, Xa, 320, 64, ea, kTwoE);
    run_mlp2(s, Xa, kTwoE, 320, ee_w1, ee_g1, ee_b1, ee_w2, ee_g2, ee_b2,
             Xi, G1, G2, Wp, stat, Xc, 128, 0, nullptr);               // e_out1 -> Xmix[:, 0:64]

    // Edge_Cycle: cycle-side second gather11 -> cyc_new f32 [RC, 320]
    g11_scatter(g11, ecT, nullptr, c5a, c5t, kR5, kC5, 128);
    g11_gather_f32(g11, cycnew, 320, 0, c5a, kR5);
    g11_scatter(g11, ecT + (size_t)kR5 * 128, nullptr, c6a, c6t, kR6, kC6, 128);
    g11_gather_f32(g11, cycnew + (size_t)kR5 * 320, 320, 0, c6a, kR6);
    k_gath_f32<<<nblk(kRC * 64), TPB, 0, s>>>(cycnew, 320, 256, cyc, nullptr, kRC, 64);

    // X_ecC (bf16 copy of cyc_new)
    k_gath_bf16<<<nblk(kRC * 320), TPB, 0, s>>>(Xa, 320, 0, cycnew, nullptr, kRC, 320);

    // c2e = gather11(cyc_new, cyca, cyct, C5+C6, ea) -> X_ecE = [er | c2e] bf16 [2E,704]
    g11_scatter(g11, cycnew, nullptr, cyca, cyct, kRC, kC5 + kC6, 320);
    k_gath_bf16<<<nblk(kTwoE * 64), TPB, 0, s>>>(Xb, 704, 0, er, nullptr, kTwoE, 64);
    g11_gather_bf16(g11, Xb, 704, 64, ea, kTwoE);

    run_mlp2(s, Xb, kTwoE, 704, ecE_w1, ecE_g1, ecE_b1, ecE_w2, ecE_g2, ecE_b2,
             Xi, G1, G2, Wp, stat, Xc, 128, 64, nullptr);              // e_out2 -> Xmix[:, 64:128]
    run_mlp2(s, Xa, kRC, 320, ecC_w1, ecC_g1, ecC_b1, ecC_w2, ecC_g2, ecC_b2,
             Xi, G1, G2, Wp, stat, nullptr, 0, 0, cyc);                // new cyc (f32)

    // mix: er = bn_relu(Xmix @ mix_w)
    run_gemm(s, Xc, kTwoE, 128, 64, mix_w, Wp, G2);
    k_colstats<<<64, TPB, 0, s>>>(G2, kTwoE, 64, stat, stat + 256);
    k_bn_relu_f32<<<nblk(kTwoE * 64), TPB, 0, s>>>(er, G2, stat, stat + 256, mix_g, mix_b, kTwoE, 64);
  }

  // ---- pooling + final head ----
  k_zero_f32<<<nblk(kN * 64), TPB, 0, s>>>(tmpL, kN * 64);
  k_zero_f32<<<nblk(kN * 64), TPB, 0, s>>>(tmpB, kN * 64);
  k_scat<<<nblk(kTwoE * 64), TPB, 0, s>>>(tmpL, 64, 0, er, ea, nullptr, kTwoE, 64);   // node1
  k_scat<<<nblk(kRC * 64), TPB, 0, s>>>(tmpB, 64, 0, cyc, cyca, nullptr, kRC, 64);    // node2
  k_zero_f32<<<nblk(512), TPB, 0, s>>>(cnt, 512);
  k_zero_f32<<<nblk(512 * 128), TPB, 0, s>>>(gp, 512 * 128);
  k_count_batch<<<nblk(kN), TPB, 0, s>>>(batch, cnt, kN);
  k_scat<<<nblk(kN * 64), TPB, 0, s>>>(gp, 128, 0,  tmpL, batch, nullptr, kN, 64);
  k_scat<<<nblk(kN * 64), TPB, 0, s>>>(gp, 128, 64, tmpB, batch, nullptr, kN, 64);

  k_zero_u16<<<nblk(kBPad * 128), TPB, 0, s>>>(Xi, kBPad * 128);
  k_pool_div_bf16<<<nblk(kB * 128), TPB, 0, s>>>(gp, cnt, Xi, kB, 128);

  run_gemm(s, Xi, kBPad, 128, 64, fin_w1, Wp, G1);
  k_colstats<<<64, TPB, 0, s>>>(G1, kB, 64, stat, stat + 256);
  k_zero_u16<<<nblk(kBPad * 64), TPB, 0, s>>>(Xc, kBPad * 64);
  k_bn_relu_bf16<<<nblk(kB * 64), TPB, 0, s>>>(Xc, 64, 0, G1, stat, stat + 256, fin_g1, fin_bb1, kB, 64);

  run_gemm(s, Xc, kBPad, 64, 64, fin_w2, Wp, G2);
  k_colstats<<<64, TPB, 0, s>>>(G2, kB, 64, stat, stat + 256);
  k_bn_relu_f32<<<nblk(kB * 64), TPB, 0, s>>>(G1, G2, stat, stat + 256, fin_g2, fin_bb2, kB, 64);

  k_outhead<<<nblk(kB), TPB, 0, s>>>(G1, out_w, out_b, (float*)d_out, kB, 64);
}
